// PerformerModel_45981919871442
// MI455X (gfx1250) — matrix operand order repaired and checked by reading, not compiled, not run on hardware
//
#include <hip/hip_runtime.h>
#include <cstdint>

// ---------------- model dims ----------------
#define Bc      4
#define SEQc    2048
#define TB      8192          // B*SEQ
#define DIMc    768
#define INNERc  768
#define HEADSc  12
#define DHc     64
#define Mf      256           // FAVOR features
#define FFc     3072
#define DEPTHc  6
#define BHc     48            // B*HEADS
#define LABELSc 2

typedef __attribute__((ext_vector_type(16))) __bf16 v16bf;
typedef __attribute__((ext_vector_type(8)))  float  v8f;
typedef unsigned int tdm_u32x4 __attribute__((ext_vector_type(4)));
typedef int          tdm_i32x8 __attribute__((ext_vector_type(8)));
typedef int          tdm_i32x4 __attribute__((ext_vector_type(4)));

struct FragBits { uint4 lo, hi; };

__device__ __forceinline__ unsigned short f2bf(float f){
  unsigned u = __float_as_uint(f);
  u += 0x7FFFu + ((u >> 16) & 1u);          // round-to-nearest-even
  return (unsigned short)(u >> 16);
}
__device__ __forceinline__ float bf2f(unsigned short s){
  return __uint_as_float(((unsigned)s) << 16);
}

// A fragment (16x32 bf16): lane l holds row (l&15); K-halves split per ISA
// layout: lanes 0-15 K=[0..7]+[16..23], lanes 16-31 K=[8..15]+[24..31].
__device__ __forceinline__ v16bf frag_a(const unsigned short* s, int row, int ld){
  int lane = threadIdx.x & 31;
  const unsigned short* p = s + (row + (lane & 15)) * ld + ((lane >> 4) << 3);
  FragBits b;
  b.lo = *(const uint4*)p;
  b.hi = *(const uint4*)(p + 16);
  return __builtin_bit_cast(v16bf, b);
}
// B fragment (32x16 bf16) read from an LDS tile stored transposed [N][K]:
// lane l holds col (l&15); lanes 0-15 K=[0..7]+[16..23], lanes 16-31 K=[8..15]+[24..31].
__device__ __forceinline__ v16bf frag_b(const unsigned short* s, int col, int ld){
  int lane = threadIdx.x & 31;
  const unsigned short* p = s + (col + (lane & 15)) * ld + ((lane >> 4) << 3);
  FragBits b;
  b.lo = *(const uint4*)p;
  b.hi = *(const uint4*)(p + 16);
  return __builtin_bit_cast(v16bf, b);
}
__device__ __forceinline__ v8f wmma_bf16(v16bf a, v16bf b, v8f c){
  return __builtin_amdgcn_wmma_f32_16x16x32_bf16(false, a, false, b, (short)0, c, false, false);
}
__device__ __forceinline__ void scat2(unsigned short* p, int stride, unsigned w){
  p[0] = (unsigned short)w; p[stride] = (unsigned short)(w >> 16);
}
__device__ __forceinline__ void scat8(unsigned short* p, int stride, uint4 w){
  scat2(p,            stride, w.x);
  scat2(p + 2*stride, stride, w.y);
  scat2(p + 4*stride, stride, w.z);
  scat2(p + 6*stride, stride, w.w);
}
__device__ __forceinline__ void atomicMaxF(float* addr, float v){
  unsigned* ua = (unsigned*)addr;
  unsigned old = *ua;
  while (__uint_as_float(old) < v){
    unsigned assumed = old;
    old = atomicCAS(ua, assumed, __float_as_uint(v));
    if (old == assumed) break;
  }
}

// ---------------------------------------------------------------------
// Tensor Data Mover: async DMA of a 2D bf16 tile (tile_rows x tile_cols,
// row stride src_stride elements) from global memory into LDS (row-major,
// contiguous). D# layout per CDNA5 ISA ch.8: group0 = {count/flags, lds,
// global_addr, type=2}; group1 = {data_size=2B, dims/tiles/strides};
// groups 2/3 zero (2D). Issued once per wave; completion via TENSORcnt.
// This toolchain exposes the 6-arg builtin (extra i32x8 group + cpol).
// ---------------------------------------------------------------------
__device__ __forceinline__ void tdm_load_2d(unsigned lds_off, const void* gptr,
                                            int tile_cols, int tile_rows,
                                            int src_stride)
{
  unsigned long long ga = (unsigned long long)(uintptr_t)gptr;
  tdm_u32x4 g0;
  g0.x = 1u;                                             // count=1, user desc
  g0.y = lds_off;                                        // lds_addr (bytes)
  g0.z = (unsigned)ga;                                   // global_addr[31:0]
  g0.w = (unsigned)((ga >> 32) & 0x1FFFFFFull) | (2u << 30);  // addr hi | type=2
  tdm_i32x8 g1;
  g1[0] = 0x00010000;                                    // data_size=1 (2 bytes)
  g1[1] = (src_stride & 0xFFFF) << 16;                   // tensor_dim0 lo16
  g1[2] = ((unsigned)src_stride >> 16) | (0x4000 << 16); // dim0 hi16 | dim1 lo16
  g1[3] = (tile_cols << 16);                             // dim1 hi16=0 | tile_dim0
  g1[4] = tile_rows;                                     // tile_dim1 | tile_dim2=0
  g1[5] = src_stride;                                    // tensor_dim0_stride lo32
  g1[6] = 0;                                             // stride0 hi | stride1 lo
  g1[7] = 0;                                             // stride1 hi
  tdm_i32x4 z4 = {0, 0, 0, 0};
  tdm_i32x8 z8 = {0, 0, 0, 0, 0, 0, 0, 0};
  __builtin_amdgcn_tensor_load_to_lds(g0, g1, z4, z4, z8, 0);
}

// =====================================================================
// Generic bf16 GEMM: C[MxN] = act(A[MxK]*B[KxN] + bias) (+residual)
// block tile 128x128x32, 8 waves (2M x 4N), wave tile 64x32 (4x2 WMMA).
// Double-buffered: A tiles staged by the Tensor Data Mover (async DMA,
// TENSORcnt), B tiles register-prefetched + transposed into LDS, both
// overlapped with the WMMA work on the current tile.
// =====================================================================
template<int ACT, bool OBF16, bool BIAS, bool RES>
__global__ __launch_bounds__(256) void gemm_bf16_kernel(
    const unsigned short* __restrict__ A, int lda,
    const unsigned short* __restrict__ Bw, int ldb,
    void* __restrict__ Cout, int ldc,
    const float* __restrict__ bias,
    const float* __restrict__ resid,
    int K)
{
  __shared__ __align__(16) unsigned short As[2][128*32];
  __shared__ __align__(16) unsigned short Bs[2][128*32];
  const int tid = threadIdx.x, lane = tid & 31, wave = tid >> 5;
  const int wm = wave & 1, wn = wave >> 1;
  const int bm = blockIdx.y * 128, bn = blockIdx.x * 128;
  v8f zero = {};
  v8f acc[4][2];
  for (int i = 0; i < 4; i++) for (int j = 0; j < 2; j++) acc[i][j] = zero;
  const int bk = tid >> 3, bn0 = (tid & 7) * 16;

  // ---- prologue: stage tile 0 into buffer 0 ----
  {
    const uint4* gb = (const uint4*)(Bw + (size_t)bk * ldb + bn + bn0);
    uint4 b0 = gb[0], b1 = gb[1];
    scat8(&Bs[0][(bn0    )*32 + bk], 32, b0);
    scat8(&Bs[0][(bn0 + 8)*32 + bk], 32, b1);
    if (wave == 0){
      tdm_load_2d((unsigned)(uintptr_t)&As[0][0],
                  A + (size_t)bm * lda, 32, 128, lda);
      __builtin_amdgcn_s_wait_tensorcnt(0);
    }
    __syncthreads();
  }

  int cur = 0;
  for (int kt = 0; kt < K; kt += 32){
    const int  nxt  = cur ^ 1;
    const bool more = (kt + 32) < K;
    uint4 nb0, nb1;
    if (more){
      // register-prefetch next B tile; kick off async DMA for next A tile
      const uint4* gb = (const uint4*)(Bw + (size_t)(kt + 32 + bk) * ldb + bn + bn0);
      nb0 = gb[0]; nb1 = gb[1];
      if (wave == 0)
        tdm_load_2d((unsigned)(uintptr_t)&As[nxt][0],
                    A + (size_t)bm * lda + kt + 32, 32, 128, lda);
    }
    // ---- compute on current tile ----
    v16bf af[4], bf[2];
    for (int i = 0; i < 4; i++) af[i] = frag_a(&As[cur][0], wm*64 + i*16, 32);
    for (int j = 0; j < 2; j++) bf[j] = frag_b(&Bs[cur][0], wn*32 + j*16, 32);
    for (int i = 0; i < 4; i++)
      for (int j = 0; j < 2; j++)
        acc[i][j] = wmma_bf16(af[i], bf[j], acc[i][j]);
    // ---- finish staging next tile ----
    if (more){
      scat8(&Bs[nxt][(bn0    )*32 + bk], 32, nb0);
      scat8(&Bs[nxt][(bn0 + 8)*32 + bk], 32, nb1);
      if (wave == 0) __builtin_amdgcn_s_wait_tensorcnt(0);
    }
    __syncthreads();
    cur = nxt;
  }

  const int half = (lane >> 4) << 3;
  for (int i = 0; i < 4; i++){
    int rowb = bm + wm*64 + i*16 + half;
    for (int j = 0; j < 2; j++){
      int col = bn + wn*32 + j*16 + (lane & 15);
      float bv = BIAS ? bias[col] : 0.0f;
      for (int r = 0; r < 8; r++){
        int row = rowb + r;
        float v = acc[i][j][r] + bv;
        if (ACT == 1) v = 0.5f * v * (1.0f + erff(v * 0.70710678118f));  // exact GELU
        if (RES) v += resid[(size_t)row * ldc + col];
        if (OBF16) ((unsigned short*)Cout)[(size_t)row * ldc + col] = f2bf(v);
        else       ((float*)Cout)[(size_t)row * ldc + col] = v;
      }
    }
  }
}

// =====================================================================
// FAVOR+ feature kernel: xd = (x*dn) @ projT, per head.
// block tile 64(rows) x 256(all features) x K=64; 8 waves (4M x 2N),
// wave tile 16x128 (8 WMMA). MODE 0: query (row-max stab, emit qp),
// MODE 1: key pass1 (global head max), MODE 2: key pass2 (emit kp).
// =====================================================================
template<int MODE>
__global__ __launch_bounds__(256) void favor_kernel(
    const unsigned short* __restrict__ X,    // [TB, INNER] bf16 (q or k)
    const unsigned short* __restrict__ PT,   // [64, 256] bf16 projT (layer)
    const float* __restrict__ diag,          // [TB, HEADS]
    float* __restrict__ headmax,             // [BH]
    unsigned short* __restrict__ outp)       // [BH, SEQ, Mf] bf16
{
  __shared__ __align__(16) unsigned short As[64*32];
  __shared__ __align__(16) unsigned short Bs[256*32];
  __shared__ float smax[64][2];
  const int tid = threadIdx.x, lane = tid & 31, wave = tid >> 5;
  const int wm = wave & 3, wn = wave >> 2;
  const int h = blockIdx.y, t0 = blockIdx.x * 64;
  const int bh = (t0 >> 11) * HEADSc + h;
  v8f zero = {};
  v8f acc[8];
  for (int j = 0; j < 8; j++) acc[j] = zero;
  const int ar = tid >> 2, ac = (tid & 3) * 8;
  const int bk = tid >> 3, bn0 = (tid & 7) * 32;

  for (int kt = 0; kt < DHc; kt += 32){
    *(uint4*)&As[ar*32 + ac] =
        *(const uint4*)(X + (size_t)(t0 + ar) * INNERc + h*DHc + kt + ac);
    const uint4* gb = (const uint4*)(PT + (size_t)(kt + bk) * Mf + bn0);
    uint4 b0 = gb[0], b1 = gb[1], b2 = gb[2], b3 = gb[3];
    scat8(&Bs[(bn0     )*32 + bk], 32, b0);
    scat8(&Bs[(bn0 +  8)*32 + bk], 32, b1);
    scat8(&Bs[(bn0 + 16)*32 + bk], 32, b2);
    scat8(&Bs[(bn0 + 24)*32 + bk], 32, b3);
    __syncthreads();
    v16bf af = frag_a(As, wm*16, 32);
    for (int j = 0; j < 8; j++){
      v16bf bfj = frag_b(Bs, wn*128 + j*16, 32);
      acc[j] = wmma_bf16(af, bfj, acc[j]);
    }
    __syncthreads();
  }

  const float dnm = 0.35355339059327373f;    // 64^-0.25
  for (int j = 0; j < 8; j++)
    for (int r = 0; r < 8; r++) acc[j][r] *= dnm;

  const int half = (lane >> 4) << 3;

  if (MODE == 1){
    float wmax = -3.4e38f;
    for (int j = 0; j < 8; j++)
      for (int r = 0; r < 8; r++) wmax = fmaxf(wmax, acc[j][r]);
    for (int mm = 1; mm < 32; mm <<= 1) wmax = fmaxf(wmax, __shfl_xor(wmax, mm, 32));
    if (lane == 0) atomicMaxF(&headmax[bh], wmax);
    return;
  }

  float stab[8], dg[8];
  if (MODE == 0){
    float rmax[8];
    for (int r = 0; r < 8; r++){
      float v = -3.4e38f;
      for (int j = 0; j < 8; j++) v = fmaxf(v, acc[j][r]);
      for (int mm = 1; mm < 16; mm <<= 1) v = fmaxf(v, __shfl_xor(v, mm, 32));
      rmax[r] = v;
    }
    if ((lane & 15) == 0)
      for (int r = 0; r < 8; r++) smax[wm*16 + half + r][wn] = rmax[r];
    __syncthreads();
    for (int r = 0; r < 8; r++){
      int rl = wm*16 + half + r;
      stab[r] = fmaxf(smax[rl][0], smax[rl][1]);
      dg[r]   = diag[(size_t)(t0 + rl) * HEADSc + h];
    }
  } else {
    float hm = headmax[bh];
    for (int r = 0; r < 8; r++){
      int rl = wm*16 + half + r;
      stab[r] = hm;
      dg[r]   = diag[(size_t)(t0 + rl) * HEADSc + h];
    }
  }

  const float ratio = 0.0625f;               // 256^-0.5
  for (int j = 0; j < 8; j++){
    int m = wn*128 + j*16 + (lane & 15);
    for (int r = 0; r < 8; r++){
      int rl = wm*16 + half + r;
      int n  = (t0 + rl) & (SEQc - 1);
      float e = ratio * (__expf(acc[j][r] - dg[r] - stab[r]) + 1e-4f);
      outp[((size_t)bh * SEQc + n) * Mf + m] = f2bf(e);
    }
  }
}

// =====================================================================
// ctx[bh,m,d] = sum_n kp[bh,n,m] * v[b,n,h*64+d]   (256x64, K=2048)
// one block per (b,h); 8 waves (4M x 2N), wave tile 64x32
// =====================================================================
__global__ __launch_bounds__(256) void ctx_kernel(
    const unsigned short* __restrict__ kp,   // [BH, SEQ, Mf]
    const unsigned short* __restrict__ V,    // [TB, INNER]
    unsigned short* __restrict__ ctx)        // [BH, Mf, DHc] bf16
{
  __shared__ __align__(16) unsigned short Ks[256*32];
  __shared__ __align__(16) unsigned short Vs[64*32];
  const int tid = threadIdx.x, lane = tid & 31, wave = tid >> 5;
  const int wm = wave & 3, wn = wave >> 2;
  const int bh = blockIdx.x, b = bh / HEADSc, h = bh % HEADSc;
  v8f zero = {};
  v8f acc[4][2];
  for (int i = 0; i < 4; i++) for (int j = 0; j < 2; j++) acc[i][j] = zero;
  const int kn = tid >> 3, m0 = (tid & 7) * 32, d0 = (tid & 7) * 8;

  for (int n0 = 0; n0 < SEQc; n0 += 32){
    const uint4* gk = (const uint4*)(kp + ((size_t)bh * SEQc + n0 + kn) * Mf + m0);
    uint4 k0 = gk[0], k1 = gk[1], k2 = gk[2], k3 = gk[3];
    scat8(&Ks[(m0     )*32 + kn], 32, k0);
    scat8(&Ks[(m0 +  8)*32 + kn], 32, k1);
    scat8(&Ks[(m0 + 16)*32 + kn], 32, k2);
    scat8(&Ks[(m0 + 24)*32 + kn], 32, k3);
    uint4 v0 = *(const uint4*)(V + ((size_t)(b * SEQc) + n0 + kn) * INNERc + h*DHc + d0);
    scat8(&Vs[d0*32 + kn], 32, v0);
    __syncthreads();
    v16bf af[4], bfv[2];
    for (int i = 0; i < 4; i++) af[i]  = frag_a(Ks, wm*64 + i*16, 32);
    for (int j = 0; j < 2; j++) bfv[j] = frag_b(Vs, wn*32 + j*16, 32);
    for (int i = 0; i < 4; i++)
      for (int j = 0; j < 2; j++)
        acc[i][j] = wmma_bf16(af[i], bfv[j], acc[i][j]);
    __syncthreads();
  }

  const int half = (lane >> 4) << 3;
  for (int i = 0; i < 4; i++)
    for (int j = 0; j < 2; j++){
      int m = wm*64 + i*16 + half, d = wn*32 + j*16 + (lane & 15);
      for (int r = 0; r < 8; r++)
        ctx[((size_t)bh * Mf + m + r) * DHc + d] = f2bf(acc[i][j][r]);
    }
}

// =====================================================================
// out[b,n,h*64+d] = d_inv[bh,n] * sum_m qp[bh,n,m] * ctx[bh,m,d]
// grid (SEQ/128, BH); 8 waves 8Mx1N, wave tile 16x64
// =====================================================================
__global__ __launch_bounds__(256) void attnout_kernel(
    const unsigned short* __restrict__ qp,   // [BH, SEQ, Mf]
    const unsigned short* __restrict__ ctx,  // [BH, Mf, DHc]
    const float* __restrict__ dinv,          // [BH, SEQ]
    unsigned short* __restrict__ outp)       // [TB, INNER] bf16
{
  __shared__ __align__(16) unsigned short As[128*32];
  __shared__ __align__(16) unsigned short Bs[64*32];
  const int tid = threadIdx.x, lane = tid & 31, wave = tid >> 5;
  const int bh = blockIdx.y, b = bh / HEADSc, h = bh % HEADSc;
  const int n0b = blockIdx.x * 128;
  v8f zero = {};
  v8f acc[4];
  for (int j = 0; j < 4; j++) acc[j] = zero;
  const int ar = tid >> 1, ac = (tid & 1) * 16;
  const int bk = tid >> 3, d0 = (tid & 7) * 8;

  for (int kt = 0; kt < Mf; kt += 32){
    const uint4* ga = (const uint4*)(qp + ((size_t)bh * SEQc + n0b + ar) * Mf + kt + ac);
    uint4 a0 = ga[0], a1 = ga[1];
    *(uint4*)&As[ar*32 + ac]     = a0;
    *(uint4*)&As[ar*32 + ac + 8] = a1;
    uint4 b0 = *(const uint4*)(ctx + (size_t)bh * Mf * DHc + (kt + bk) * DHc + d0);
    scat8(&Bs[d0*32 + bk], 32, b0);
    __syncthreads();
    v16bf af = frag_a(As, wave*16, 32);
    for (int j = 0; j < 4; j++){
      v16bf bfj = frag_b(Bs, j*16, 32);
      acc[j] = wmma_bf16(af, bfj, acc[j]);
    }
    __syncthreads();
  }

  const int half = (lane >> 4) << 3;
  for (int r = 0; r < 8; r++){
    int n = n0b + wave*16 + half + r;
    float sc = dinv[(size_t)bh * SEQc + n];
    for (int j = 0; j < 4; j++){
      int d = j*16 + (lane & 15);
      outp[((size_t)(b * SEQc) + n) * INNERc + h*DHc + d] = f2bf(acc[j][r] * sc);
    }
  }
}

// ---------------- small scalar kernels ----------------
__global__ __launch_bounds__(256) void layernorm_kernel(
    const float* __restrict__ X, const float* __restrict__ g,
    const float* __restrict__ bta, unsigned short* __restrict__ Y)
{
  __shared__ float rs[256], rs2[256];
  const int row = blockIdx.x, tid = threadIdx.x;
  float xv[3], s = 0.f, s2 = 0.f;
  for (int i = 0; i < 3; i++){
    float v = X[(size_t)row * DIMc + tid + i*256];
    xv[i] = v; s += v; s2 += v*v;
  }
  rs[tid] = s; rs2[tid] = s2; __syncthreads();
  for (int st = 128; st > 0; st >>= 1){
    if (tid < st){ rs[tid] += rs[tid+st]; rs2[tid] += rs2[tid+st]; }
    __syncthreads();
  }
  float mu = rs[0] * (1.0f/DIMc);
  float var = rs2[0] * (1.0f/DIMc) - mu*mu;
  float rstd = rsqrtf(var + 1e-5f);
  for (int i = 0; i < 3; i++){
    int c = tid + i*256;
    Y[(size_t)row * DIMc + c] = f2bf((xv[i]-mu)*rstd*g[c] + bta[c]);
  }
}

__global__ __launch_bounds__(256) void diag_kernel(
    const unsigned short* __restrict__ X, float* __restrict__ dg)
{
  int i = blockIdx.x*256 + threadIdx.x;     // TB*HEADS
  int t = i / HEADSc, h = i - t*HEADSc;
  const unsigned short* p = X + (size_t)t * INNERc + h*DHc;
  float s = 0.f;
  for (int d = 0; d < DHc; d++){ float v = bf2f(p[d]); s += v*v; }
  dg[i] = 0.0625f * s;                      // 0.5 * dn^2 = 0.5/8
}

__global__ __launch_bounds__(256) void ksum_kernel(
    const unsigned short* __restrict__ kp, float* __restrict__ ks)
{
  int bh = blockIdx.x, m = threadIdx.x;     // 256 threads == Mf
  float s = 0.f;
  const unsigned short* p = kp + (size_t)bh * SEQc * Mf + m;
  for (int n = 0; n < SEQc; n++) s += bf2f(p[(size_t)n * Mf]);
  ks[bh*Mf + m] = s;
}

__global__ __launch_bounds__(256) void dinv_kernel(
    const unsigned short* __restrict__ qp, const float* __restrict__ ks,
    float* __restrict__ di)
{
  int i = blockIdx.x*256 + threadIdx.x;     // BH*SEQ
  int bh = i >> 11, n = i & (SEQc - 1);
  const unsigned short* p = qp + ((size_t)bh * SEQc + n) * Mf;
  const float* k = ks + bh * Mf;
  float s = 0.f;
  for (int m = 0; m < Mf; m++) s += bf2f(p[m]) * k[m];
  di[i] = 1.0f / s;
}

__global__ void initmax_kernel(float* hm){ if (threadIdx.x < BHc) hm[threadIdx.x] = -3.4e38f; }

__global__ __launch_bounds__(256) void embed_kernel(
    const int* __restrict__ x, const float* __restrict__ te,
    const float* __restrict__ pe, float* __restrict__ h)
{
  int i = blockIdx.x*256 + threadIdx.x;     // TB*DIM
  int t = i / DIMc, d = i - t*DIMc;
  int tok = x[t];
  h[i] = te[(size_t)tok * DIMc + d] + pe[(size_t)(t & (SEQc-1)) * DIMc + d];
}

__global__ __launch_bounds__(256) void cvt_bf16_kernel(
    const float* __restrict__ s, unsigned short* __restrict__ d, int n)
{
  int i = blockIdx.x*256 + threadIdx.x;
  if (i < n) d[i] = f2bf(s[i]);
}

__global__ __launch_bounds__(256) void projT_kernel(
    const float* __restrict__ p, unsigned short* __restrict__ pt)
{
  int i = blockIdx.x*256 + threadIdx.x;     // DEPTH*DH*Mf
  int l = i / (DHc*Mf); int rem = i - l*DHc*Mf;
  int d = rem / Mf; int m = rem - d*Mf;
  pt[i] = f2bf(p[((size_t)l*Mf + m)*DHc + d]);
}

// final LN (first token only) + pooler tanh + classifier
__global__ __launch_bounds__(256) void head_kernel(
    const float* __restrict__ H, const float* __restrict__ lg, const float* __restrict__ lb,
    const float* __restrict__ Wp, const float* __restrict__ bp,
    const float* __restrict__ Wc, const float* __restrict__ bc,
    float* __restrict__ out)
{
  __shared__ float nrm[DIMc], pooled[DIMc];
  __shared__ float rs[256], rs2[256];
  const int b = blockIdx.x, tid = threadIdx.x;
  const float* xr = H + (size_t)b * SEQc * DIMc;     // token 0 of batch b
  float xv[3], s = 0.f, s2 = 0.f;
  for (int i = 0; i < 3; i++){ float v = xr[tid + i*256]; xv[i]=v; s+=v; s2+=v*v; }
  rs[tid]=s; rs2[tid]=s2; __syncthreads();
  for (int st = 128; st > 0; st >>= 1){
    if (tid < st){ rs[tid]+=rs[tid+st]; rs2[tid]+=rs2[tid+st]; }
    __syncthreads();
  }
  float mu = rs[0]*(1.0f/DIMc), var = rs2[0]*(1.0f/DIMc) - mu*mu;
  float rstd = rsqrtf(var + 1e-5f);
  for (int i = 0; i < 3; i++){ int c = tid + i*256; nrm[c] = (xv[i]-mu)*rstd*lg[c] + lb[c]; }
  __syncthreads();
  for (int j = tid; j < DIMc; j += 256){
    float a = 0.f;
    for (int i2 = 0; i2 < DIMc; i2++) a += nrm[i2] * Wp[(size_t)i2*DIMc + j];
    pooled[j] = tanhf(a + bp[j]);
  }
  __syncthreads();
  for (int l = 0; l < LABELSc; l++){
    float a = 0.f;
    for (int i2 = tid; i2 < DIMc; i2 += 256) a += pooled[i2] * Wc[(size_t)i2*LABELSc + l];
    rs[tid] = a; __syncthreads();
    for (int st = 128; st > 0; st >>= 1){ if (tid < st) rs[tid] += rs[tid+st]; __syncthreads(); }
    if (tid == 0) out[b*LABELSc + l] = rs[0] + bc[l];
    __syncthreads();
  }
}

// =====================================================================
extern "C" void kernel_launch(void* const* d_in, const int* in_sizes, int n_in,
                              void* d_out, int out_size, void* d_ws, size_t ws_size,
                              hipStream_t stream)
{
  (void)in_sizes; (void)n_in; (void)out_size; (void)ws_size;
  const int*   x    = (const int*)  d_in[0];
  const float* te   = (const float*)d_in[1];
  const float* pe   = (const float*)d_in[2];
  const float* ln1g = (const float*)d_in[3];
  const float* ln1b = (const float*)d_in[4];
  const float* Wq   = (const float*)d_in[5];
  const float* Wk   = (const float*)d_in[6];
  const float* Wv   = (const float*)d_in[7];
  const float* Wo   = (const float*)d_in[8];
  const float* bo   = (const float*)d_in[9];
  const float* proj = (const float*)d_in[10];
  const float* ln2g = (const float*)d_in[11];
  const float* ln2b = (const float*)d_in[12];
  const float* W1   = (const float*)d_in[13];
  const float* b1   = (const float*)d_in[14];
  const float* W2   = (const float*)d_in[15];
  const float* b2   = (const float*)d_in[16];
  const float* lnfg = (const float*)d_in[17];
  const float* lnfb = (const float*)d_in[18];
  const float* Wp   = (const float*)d_in[19];
  const float* bp   = (const float*)d_in[20];
  const float* Wc   = (const float*)d_in[21];
  const float* bc   = (const float*)d_in[22];

  char* base = (char*)d_ws; size_t off = 0;
  auto alloc = [&](size_t bytes) -> void* {
    void* p = base + off; off = (off + bytes + 255) & ~(size_t)255; return p;
  };

  float*          h      = (float*)alloc((size_t)TB*DIMc*4);
  unsigned short* y      = (unsigned short*)alloc((size_t)TB*DIMc*2);
  unsigned short* q      = (unsigned short*)alloc((size_t)TB*INNERc*2);
  unsigned short* k      = (unsigned short*)alloc((size_t)TB*INNERc*2);
  unsigned short* v      = (unsigned short*)alloc((size_t)TB*INNERc*2);
  unsigned short* qp     = (unsigned short*)alloc((size_t)BHc*SEQc*Mf*2);
  unsigned short* kp     = (unsigned short*)alloc((size_t)BHc*SEQc*Mf*2);
  float*          diag_q = (float*)alloc((size_t)TB*HEADSc*4);
  float*          diag_k = (float*)alloc((size_t)TB*HEADSc*4);
  float*          hmax   = (float*)alloc((size_t)BHc*4);
  float*          ksum   = (float*)alloc((size_t)BHc*Mf*4);
  float*          dinvb  = (float*)alloc((size_t)BHc*SEQc*4);
  unsigned short* ctxb   = (unsigned short*)alloc((size_t)BHc*Mf*DHc*2);
  unsigned short* attn   = (unsigned short*)alloc((size_t)TB*INNERc*2);
  unsigned short* ff1    = (unsigned short*)alloc((size_t)TB*FFc*2);
  unsigned short* WqB    = (unsigned short*)alloc((size_t)DEPTHc*DIMc*INNERc*2);
  unsigned short* WkB    = (unsigned short*)alloc((size_t)DEPTHc*DIMc*INNERc*2);
  unsigned short* WvB    = (unsigned short*)alloc((size_t)DEPTHc*DIMc*INNERc*2);
  unsigned short* WoB    = (unsigned short*)alloc((size_t)DEPTHc*INNERc*DIMc*2);
  unsigned short* W1B    = (unsigned short*)alloc((size_t)DEPTHc*DIMc*FFc*2);
  unsigned short* W2B    = (unsigned short*)alloc((size_t)DEPTHc*FFc*DIMc*2);
  unsigned short* projTB = (unsigned short*)alloc((size_t)DEPTHc*DHc*Mf*2);

  // weight conversion (every call; deterministic)
  const int nW = DEPTHc*DIMc*INNERc, nF = DEPTHc*DIMc*FFc;
  cvt_bf16_kernel<<<(nW+255)/256,256,0,stream>>>(Wq, WqB, nW);
  cvt_bf16_kernel<<<(nW+255)/256,256,0,stream>>>(Wk, WkB, nW);
  cvt_bf16_kernel<<<(nW+255)/256,256,0,stream>>>(Wv, WvB, nW);
  cvt_bf16_kernel<<<(nW+255)/256,256,0,stream>>>(Wo, WoB, nW);
  cvt_bf16_kernel<<<(nF+255)/256,256,0,stream>>>(W1, W1B, nF);
  cvt_bf16_kernel<<<(nF+255)/256,256,0,stream>>>(W2, W2B, nF);
  projT_kernel<<<(DEPTHc*DHc*Mf)/256,256,0,stream>>>(proj, projTB);

  embed_kernel<<<(TB*DIMc)/256,256,0,stream>>>(x, te, pe, h);

  for (int l = 0; l < DEPTHc; l++){
    const unsigned short* WqL = WqB + (size_t)l*DIMc*INNERc;
    const unsigned short* WkL = WkB + (size_t)l*DIMc*INNERc;
    const unsigned short* WvL = WvB + (size_t)l*DIMc*INNERc;
    const unsigned short* WoL = WoB + (size_t)l*INNERc*DIMc;
    const unsigned short* W1L = W1B + (size_t)l*DIMc*FFc;
    const unsigned short* W2L = W2B + (size_t)l*FFc*DIMc;
    const unsigned short* PTL = projTB + (size_t)l*DHc*Mf;

    layernorm_kernel<<<TB,256,0,stream>>>(h, ln1g + l*DIMc, ln1b + l*DIMc, y);

    dim3 g1(INNERc/128, TB/128);
    gemm_bf16_kernel<0,true,false,false><<<g1,256,0,stream>>>(y,DIMc, WqL,INNERc, q,INNERc, nullptr,nullptr, DIMc);
    gemm_bf16_kernel<0,true,false,false><<<g1,256,0,stream>>>(y,DIMc, WkL,INNERc, k,INNERc, nullptr,nullptr, DIMc);
    gemm_bf16_kernel<0,true,false,false><<<g1,256,0,stream>>>(y,DIMc, WvL,INNERc, v,INNERc, nullptr,nullptr, DIMc);

    diag_kernel<<<(TB*HEADSc)/256,256,0,stream>>>(q, diag_q);
    diag_kernel<<<(TB*HEADSc)/256,256,0,stream>>>(k, diag_k);

    dim3 gf(TB/64, HEADSc);
    favor_kernel<0><<<gf,256,0,stream>>>(q, PTL, diag_q, nullptr, qp);
    initmax_kernel<<<1,64,0,stream>>>(hmax);
    favor_kernel<1><<<gf,256,0,stream>>>(k, PTL, nullptr, hmax, nullptr);
    favor_kernel<2><<<gf,256,0,stream>>>(k, PTL, diag_k, hmax, kp);

    ksum_kernel<<<BHc,256,0,stream>>>(kp, ksum);
    ctx_kernel<<<BHc,256,0,stream>>>(kp, v, ctxb);
    dinv_kernel<<<(BHc*SEQc)/256,256,0,stream>>>(qp, ksum, dinvb);
    dim3 go(SEQc/128, BHc);
    attnout_kernel<<<go,256,0,stream>>>(qp, ctxb, dinvb, attn);

    dim3 g2(DIMc/128, TB/128);
    gemm_bf16_kernel<0,false,true,true><<<g2,256,0,stream>>>(attn,INNERc, WoL,DIMc, h,DIMc, bo + l*DIMc, h, INNERc);

    layernorm_kernel<<<TB,256,0,stream>>>(h, ln2g + l*DIMc, ln2b + l*DIMc, y);
    dim3 g3(FFc/128, TB/128);
    gemm_bf16_kernel<1,true,true,false><<<g3,256,0,stream>>>(y,DIMc, W1L,FFc, ff1,FFc, b1 + l*FFc, nullptr, DIMc);
    dim3 g4(DIMc/128, TB/128);
    gemm_bf16_kernel<0,false,true,true><<<g4,256,0,stream>>>(ff1,FFc, W2L,DIMc, h,DIMc, b2 + l*DIMc, h, FFc);
  }

  head_kernel<<<Bc,256,0,stream>>>(h, lnfg, lnfb, Wp, bp, Wc, bc, (float*)d_out);
}
